// Block_54219667145535
// MI455X (gfx1250) — matrix operand order repaired and checked by reading, not compiled, not run on hardware
//
#include <hip/hip_runtime.h>
#include <math.h>

typedef _Float16 v16h __attribute__((ext_vector_type(16)));
typedef _Float16 v8h  __attribute__((ext_vector_type(8)));
typedef float    v8f  __attribute__((ext_vector_type(8)));
typedef int      gv4i __attribute__((vector_size(16)));   // GCC-style int4 (builtin param type)

#define DEV __device__ __forceinline__
#define AS1 __attribute__((address_space(1)))
#define AS3 __attribute__((address_space(3)))

#if __has_builtin(__builtin_amdgcn_global_load_async_to_lds_b128) && \
    __has_builtin(__builtin_amdgcn_s_wait_asynccnt)
#define HAVE_ASYNC 1
#define ASYNC_WAIT(n) __builtin_amdgcn_s_wait_asynccnt(n)
#else
#define HAVE_ASYNC 0
#define ASYNC_WAIT(n)
#endif

// ---- problem dims ----
constexpr int D      = 1024;
constexpr int S      = 4096;
constexpr int NB     = 4;            // batch
constexpr int M_TOT  = NB * S;       // 16384 rows
constexpr int H      = 2730;         // FFN hidden
constexpr int HP     = 2816;         // padded to 128-multiple (zeros)
constexpr int N_QKV  = 3 * D;        // 3072
constexpr float LN_EPS   = 1e-5f;
constexpr float ATTN_EPS = 1e-6f;

constexpr int ASZ   = 64 * 32;    // one A buffer (halfs)
constexpr int BSZ   = 128 * 32;   // one B buffer (halfs)
constexpr int TS_LD = 80;         // padded transpose-tile stride (halfs, 160B: 16B-aligned)

// 16B global -> 16B LDS copy; async-to-LDS when the toolchain has it.
DEV void copy16(const _Float16* g, _Float16* l) {
#if HAVE_ASYNC
  __builtin_amdgcn_global_load_async_to_lds_b128(
      (AS1 gv4i*)g, (AS3 gv4i*)l, 0, 0);
#else
  *(v8h*)l = *(const v8h*)g;
#endif
}

// ---- GEMM tile config: 256 threads = 8 waves, block tile 64x128, K-step 32 ----
// wave grid 2(M) x 4(N); each wave: 32x32 = 2x2 fragments of 16x16.
// Triple-buffered LDS; async stage k+2 issued in iteration k; partial
// ASYNCcnt waits (in-order retirement) leave the k+1 stage in flight.

DEV void stage_A_row(const _Float16* A, int lda, int row0, int k0,
                     _Float16* As, int tid) {
  int r  = tid >> 2;
  int kc = (tid & 3) << 3;
  copy16(A + (size_t)(row0 + r) * lda + k0 + kc, As + r * 32 + kc);
}

DEV void stage_B_rowT(const _Float16* BT, int ldk, int k0, int n0,
                      _Float16* BsT, int tid) {
#pragma unroll
  for (int i = 0; i < 2; ++i) {
    int idx = tid + i * 256;
    int n   = idx >> 2;
    int kc  = (idx & 3) << 3;
    copy16(BT + (size_t)(n0 + n) * ldk + k0 + kc, BsT + n * 32 + kc);
  }
}

DEV v16h load_a_frag(const _Float16* As, int m0, int lane) {
  // 16-bit A 16x32 layout: lanes0-15: K0-7 & K16-23 ; lanes16-31: K8-15 & K24-31
  int l  = lane & 15;
  int hi = lane >> 4;
  const v8h* p = (const v8h*)(As + (m0 + l) * 32);
  v8h x0 = p[hi];
  v8h x1 = p[hi + 2];
  v16h a;
#pragma unroll
  for (int j = 0; j < 8; ++j) { a[j] = x0[j]; a[j + 8] = x1[j]; }
  return a;
}

DEV v16h load_b_frag(const _Float16* BsT, int n0, int lane) {
  // 16-bit B 32x16 layout: lanes0-15: K0-7 & K16-23 ; lanes16-31: K8-15 & K24-31 (N=lane)
  int l = lane & 15;
  int g = lane >> 4;
  const v8h* p = (const v8h*)(BsT + (n0 + l) * 32);
  v8h x0 = p[g];
  v8h x1 = p[g + 2];
  v16h b;
#pragma unroll
  for (int j = 0; j < 8; ++j) { b[j] = x0[j]; b[j + 8] = x1[j]; }
  return b;
}

DEV v8f wmma16(v16h a, v16h b, v8f c) {
  return __builtin_amdgcn_wmma_f32_16x16x32_f16(false, a, false, b,
                                                (short)0, c, false, false);
}

// Triple-buffered GEMM core. As: [3][64*32], BsT: [3][128*32].
DEV void gemm_core(const _Float16* A, int lda, const _Float16* BT, int ldk,
                   int row0, int n0, int K,
                   _Float16* As, _Float16* BsT, v8f acc[2][2]) {
  int tid  = threadIdx.x;
  int lane = tid & 31;
  int w    = tid >> 5;
  int wm   = w & 1;
  int wn   = w >> 1;
#pragma unroll
  for (int mf = 0; mf < 2; ++mf)
#pragma unroll
    for (int nf = 0; nf < 2; ++nf) acc[mf][nf] = (v8f)(0.0f);

  // prime buffers 0 and 1
  stage_A_row(A, lda, row0, 0, As, tid);
  stage_B_rowT(BT, ldk, 0, n0, BsT, tid);
  stage_A_row(A, lda, row0, 32, As + ASZ, tid);
  stage_B_rowT(BT, ldk, 32, n0, BsT + BSZ, tid);

  int nk = K >> 5;
  for (int kt = 0; kt < nk; ++kt) {
    // 3 async ops/thread/stage; retire in order: <=3 left means stage kt landed
    if (kt + 1 < nk) { ASYNC_WAIT(3); } else { ASYNC_WAIT(0); }
    __syncthreads();
    int cur = kt % 3;
    const _Float16* Ab = As  + cur * ASZ;
    const _Float16* Bb = BsT + cur * BSZ;

    v16h a0 = load_a_frag(Ab, wm * 32, lane);
    v16h a1 = load_a_frag(Ab, wm * 32 + 16, lane);
    v16h b0 = load_b_frag(Bb, wn * 32, lane);
    v16h b1 = load_b_frag(Bb, wn * 32 + 16, lane);

    if (kt + 2 < nk) {
      int k0 = (kt + 2) << 5;
      int nb = (kt + 2) % 3;
      stage_A_row(A, lda, row0, k0, As + nb * ASZ, tid);
      stage_B_rowT(BT, ldk, k0, n0, BsT + nb * BSZ, tid);
    }

    acc[0][0] = wmma16(a0, b0, acc[0][0]);
    acc[0][1] = wmma16(a0, b1, acc[0][1]);
    acc[1][0] = wmma16(a1, b0, acc[1][0]);
    acc[1][1] = wmma16(a1, b1, acc[1][1]);
  }
}

DEV float elu1(float x) { return x > 0.0f ? x + 1.0f : __expf(x); }
DEV float silu(float x) { return x / (1.0f + __expf(-x)); }

// Transposed epilogue: scatter 64x128 tile into padded LDS once, then write
// T[col][row] with coalesced 16B stores. dst points at T[n0][row0].
DEV void epilogue_transpose(v8f acc[2][2], _Float16* Ts,
                            const float* bias, int n0, bool do_elu,
                            _Float16* dst, size_t ldT) {
  int tid = threadIdx.x, lane = tid & 31, w = tid >> 5, wm = w & 1, wn = w >> 1;
#pragma unroll
  for (int mf = 0; mf < 2; ++mf)
#pragma unroll
    for (int nf = 0; nf < 2; ++nf)
#pragma unroll
      for (int i = 0; i < 8; ++i) {
        int rl = wm * 32 + mf * 16 + (lane >> 4) * 8 + i;
        int cl = wn * 32 + nf * 16 + (lane & 15);
        float val = acc[mf][nf][i];
        if (bias) val += bias[n0 + cl];
        if (do_elu) val = elu1(val);
        Ts[cl * TS_LD + rl] = (_Float16)val;
      }
  __syncthreads();
#pragma unroll
  for (int it = 0; it < 4; ++it) {
    int idx = tid + it * 256;
    int col = idx >> 3;
    int rc  = (idx & 7) << 3;
    v8h v = *(const v8h*)(Ts + col * TS_LD + rc);
    *(v8h*)(dst + (size_t)col * ldT + rc) = v;
  }
}

// ================= conversion / padding =================
__global__ void k_cvt_f16(const float* __restrict__ s, _Float16* __restrict__ d,
                          int n) {
  for (int i = blockIdx.x * blockDim.x + threadIdx.x; i < n;
       i += gridDim.x * blockDim.x)
    d[i] = (_Float16)s[i];
}

// dst[n][k] = (n<Nv && k<Kv) ? src[k][n] : 0 ; dst is [Nd][Kd], src ld = sld
__global__ void k_transpose_pad(const float* __restrict__ s,
                                _Float16* __restrict__ d,
                                int Kd, int Nd, int Nv, int Kv, int sld) {
  int n = Nd * Kd;
  for (int i = blockIdx.x * blockDim.x + threadIdx.x; i < n;
       i += gridDim.x * blockDim.x) {
    int r = i / Kd;
    int k = i - r * Kd;
    float val = (r < Nv && k < Kv) ? s[(size_t)k * sld + r] : 0.0f;
    d[i] = (_Float16)val;
  }
}

__global__ void k_pad_bias(const float* __restrict__ s, float* __restrict__ d,
                           int sn, int dn) {
  for (int i = blockIdx.x * blockDim.x + threadIdx.x; i < dn;
       i += gridDim.x * blockDim.x)
    d[i] = (i < sn) ? s[i] : 0.0f;
}

// ================= GEMM kernels =================
// qkv: columns [0,D) -> qp row-major (elu+1); [D,2D) -> kpT transposed (elu+1);
//      [2D,3D) -> vT transposed.
__global__ void __launch_bounds__(256)
k_gemm_qkv(const _Float16* __restrict__ A, const _Float16* __restrict__ WT,
           const float* __restrict__ bqkv, _Float16* __restrict__ qp,
           _Float16* __restrict__ kpT, _Float16* __restrict__ vT) {
  __shared__ _Float16 As[3 * ASZ];
  __shared__ _Float16 BsT[3 * BSZ];
  __shared__ _Float16 Ts[128 * TS_LD];
  int row0 = blockIdx.y * 64, n0 = blockIdx.x * 128;
  v8f acc[2][2];
  gemm_core(A, D, WT, D, row0, n0, D, As, BsT, acc);

  int b  = row0 >> 12;        // batch (4096 rows each; 64-row tiles never straddle)
  int s0 = row0 & (S - 1);    // position within batch
  if (n0 < D) {
    int lane = threadIdx.x & 31, w = threadIdx.x >> 5, wm = w & 1, wn = w >> 1;
#pragma unroll
    for (int mf = 0; mf < 2; ++mf)
#pragma unroll
      for (int nf = 0; nf < 2; ++nf)
#pragma unroll
        for (int i = 0; i < 8; ++i) {
          int row = row0 + wm * 32 + mf * 16 + (lane >> 4) * 8 + i;
          int col = n0 + wn * 32 + nf * 16 + (lane & 15);
          qp[(size_t)row * D + col] = (_Float16)elu1(acc[mf][nf][i] + bqkv[col]);
        }
  } else if (n0 < 2 * D) {
    _Float16* dst = kpT + ((size_t)b * D + (n0 - D)) * S + s0;
    epilogue_transpose(acc, Ts, bqkv, n0, true, dst, S);
  } else {
    _Float16* dst = vT + ((size_t)b * D + (n0 - 2 * D)) * S + s0;
    epilogue_transpose(acc, Ts, bqkv, n0, false, dst, S);
  }
}

// kv[d][e] = sum_s kpT[d][s] * vT[e][s] ; stored transposed as kvT[e][d]
__global__ void __launch_bounds__(256)
k_gemm_kv(const _Float16* __restrict__ kpT, const _Float16* __restrict__ vT,
          _Float16* __restrict__ kvT) {
  __shared__ _Float16 As[3 * ASZ];
  __shared__ _Float16 BsT[3 * BSZ];
  __shared__ _Float16 Ts[128 * TS_LD];
  int row0 = blockIdx.y * 64, n0 = blockIdx.x * 128, b = blockIdx.z;
  const _Float16* Ab = kpT + (size_t)b * D * S;   // [D][S]
  const _Float16* Bb = vT  + (size_t)b * D * S;   // [D][S]
  v8f acc[2][2];
  gemm_core(Ab, S, Bb, S, row0, n0, S, As, BsT, acc);

  _Float16* dst = kvT + (size_t)b * D * D + (size_t)n0 * D + row0;
  epilogue_transpose(acc, Ts, nullptr, n0, false, dst, D);
}

// num = qp @ kv : A = qp [M][D], BT = kvT [D(e)][D(d)]
__global__ void __launch_bounds__(256)
k_gemm_num(const _Float16* __restrict__ qp, const _Float16* __restrict__ kvT,
           float* __restrict__ num) {
  __shared__ _Float16 As[3 * ASZ];
  __shared__ _Float16 BsT[3 * BSZ];
  int row0 = blockIdx.y * 64, n0 = blockIdx.x * 128;
  int b = row0 >> 12;
  v8f acc[2][2];
  gemm_core(qp, D, kvT + (size_t)b * D * D, D, row0, n0, D, As, BsT, acc);

  int lane = threadIdx.x & 31, w = threadIdx.x >> 5, wm = w & 1, wn = w >> 1;
#pragma unroll
  for (int mf = 0; mf < 2; ++mf)
#pragma unroll
    for (int nf = 0; nf < 2; ++nf)
#pragma unroll
      for (int i = 0; i < 8; ++i) {
        int row = row0 + wm * 32 + mf * 16 + (lane >> 4) * 8 + i;
        int col = n0 + wn * 32 + nf * 16 + (lane & 15);
        num[(size_t)row * D + col] = acc[mf][nf][i];
      }
}

__global__ void __launch_bounds__(256)
k_gemm_gateup(const _Float16* __restrict__ A, const _Float16* __restrict__ WgT,
              const _Float16* __restrict__ WuT, const float* __restrict__ bg,
              const float* __restrict__ bu, _Float16* __restrict__ h) {
  __shared__ _Float16 As[3 * ASZ];
  __shared__ _Float16 BgT[3 * BSZ];
  __shared__ _Float16 BuT[3 * BSZ];
  int row0 = blockIdx.y * 64, n0 = blockIdx.x * 128;
  int tid = threadIdx.x, lane = tid & 31, w = tid >> 5, wm = w & 1, wn = w >> 1;

  v8f ag[2][2], au[2][2];
#pragma unroll
  for (int mf = 0; mf < 2; ++mf)
#pragma unroll
    for (int nf = 0; nf < 2; ++nf) { ag[mf][nf] = (v8f)(0.0f); au[mf][nf] = (v8f)(0.0f); }

  stage_A_row(A, D, row0, 0, As, tid);
  stage_B_rowT(WgT, D, 0, n0, BgT, tid);
  stage_B_rowT(WuT, D, 0, n0, BuT, tid);
  stage_A_row(A, D, row0, 32, As + ASZ, tid);
  stage_B_rowT(WgT, D, 32, n0, BgT + BSZ, tid);
  stage_B_rowT(WuT, D, 32, n0, BuT + BSZ, tid);

  constexpr int nk = D >> 5;
  for (int kt = 0; kt < nk; ++kt) {
    // 5 async ops/thread/stage here (1 A + 2 Bg + 2 Bu)
    if (kt + 1 < nk) { ASYNC_WAIT(5); } else { ASYNC_WAIT(0); }
    __syncthreads();
    int cur = kt % 3;
    const _Float16* Ab = As  + cur * ASZ;
    const _Float16* Gb = BgT + cur * BSZ;
    const _Float16* Ub = BuT + cur * BSZ;

    v16h a0 = load_a_frag(Ab, wm * 32, lane);
    v16h a1 = load_a_frag(Ab, wm * 32 + 16, lane);
    v16h g0 = load_b_frag(Gb, wn * 32, lane);
    v16h g1 = load_b_frag(Gb, wn * 32 + 16, lane);
    v16h u0 = load_b_frag(Ub, wn * 32, lane);
    v16h u1 = load_b_frag(Ub, wn * 32 + 16, lane);

    if (kt + 2 < nk) {
      int k0 = (kt + 2) << 5;
      int nb = (kt + 2) % 3;
      stage_A_row(A, D, row0, k0, As + nb * ASZ, tid);
      stage_B_rowT(WgT, D, k0, n0, BgT + nb * BSZ, tid);
      stage_B_rowT(WuT, D, k0, n0, BuT + nb * BSZ, tid);
    }

    ag[0][0] = wmma16(a0, g0, ag[0][0]);  ag[0][1] = wmma16(a0, g1, ag[0][1]);
    ag[1][0] = wmma16(a1, g0, ag[1][0]);  ag[1][1] = wmma16(a1, g1, ag[1][1]);
    au[0][0] = wmma16(a0, u0, au[0][0]);  au[0][1] = wmma16(a0, u1, au[0][1]);
    au[1][0] = wmma16(a1, u0, au[1][0]);  au[1][1] = wmma16(a1, u1, au[1][1]);
  }

#pragma unroll
  for (int mf = 0; mf < 2; ++mf)
#pragma unroll
    for (int nf = 0; nf < 2; ++nf)
#pragma unroll
      for (int i = 0; i < 8; ++i) {
        int row = row0 + wm * 32 + mf * 16 + (lane >> 4) * 8 + i;
        int col = n0 + wn * 32 + nf * 16 + (lane & 15);
        float g = ag[mf][nf][i] + bg[col];
        float u = au[mf][nf][i] + bu[col];
        h[(size_t)row * HP + col] = (_Float16)(silu(g) * u);
      }
}

__global__ void __launch_bounds__(256)
k_gemm_down(const _Float16* __restrict__ h, const _Float16* __restrict__ WdT,
            const float* __restrict__ bd, float* __restrict__ ffn) {
  __shared__ _Float16 As[3 * ASZ];
  __shared__ _Float16 BsT[3 * BSZ];
  int row0 = blockIdx.y * 64, n0 = blockIdx.x * 128;
  v8f acc[2][2];
  gemm_core(h, HP, WdT, HP, row0, n0, HP, As, BsT, acc);

  int lane = threadIdx.x & 31, w = threadIdx.x >> 5, wm = w & 1, wn = w >> 1;
#pragma unroll
  for (int mf = 0; mf < 2; ++mf)
#pragma unroll
    for (int nf = 0; nf < 2; ++nf)
#pragma unroll
      for (int i = 0; i < 8; ++i) {
        int row = row0 + wm * 32 + mf * 16 + (lane >> 4) * 8 + i;
        int col = n0 + wn * 32 + nf * 16 + (lane & 15);
        ffn[(size_t)row * D + col] = acc[mf][nf][i] + bd[col];
      }
}

// ================= reductions & pointwise =================
// k_sum[b][d] = sum_s kpT[b][d][s]  (rows are contiguous)
__global__ void k_ksum(const _Float16* __restrict__ kpT, float* __restrict__ ks) {
  int w = threadIdx.x >> 5, lane = threadIdx.x & 31;
  int row = blockIdx.x * 8 + w;   // over NB*D
  const _Float16* p = kpT + (size_t)row * S;
  float s = 0.0f;
  for (int i = 0; i < S / 32; ++i) s += (float)p[i * 32 + lane];
#pragma unroll
  for (int o = 16; o > 0; o >>= 1) s += __shfl_xor(s, o, 32);
  if (lane == 0) ks[row] = s;
}

__global__ void k_den(const _Float16* __restrict__ qp,
                      const float* __restrict__ ks, float* __restrict__ den) {
  int w = threadIdx.x >> 5, lane = threadIdx.x & 31;
  int row = blockIdx.x * 8 + w;
  int b = row >> 12;
  const _Float16* q = qp + (size_t)row * D;
  const float* k = ks + (size_t)b * D;
  float s = 0.0f;
  for (int i = 0; i < D / 32; ++i) {
    int n = i * 32 + lane;
    s += (float)q[n] * k[n];
  }
#pragma unroll
  for (int o = 16; o > 0; o >>= 1) s += __shfl_xor(s, o, 32);
  if (lane == 0) den[row] = s;
}

// y = resid + add (optionally add scaled by 1/(den+eps)); layernorm(y)*g+b
__global__ void __launch_bounds__(256)
k_resid_ln(const float* __restrict__ resid, const float* __restrict__ add,
           const float* __restrict__ den,  // null -> plain add
           const float* __restrict__ gamma, const float* __restrict__ beta,
           float* __restrict__ out_f32, _Float16* __restrict__ out_f16) {
  __shared__ float rbuf[16];
  __shared__ float s_mu, s_rstd;
  int r = blockIdx.x, tid = threadIdx.x;
  int lane = tid & 31, w = tid >> 5;
  float scale = 1.0f;
  if (den) scale = 1.0f / (den[r] + ATTN_EPS);

  float y[4];
  float s1 = 0.0f, s2 = 0.0f;
#pragma unroll
  for (int i = 0; i < 4; ++i) {
    int n = tid + i * 256;
    float v = resid[(size_t)r * D + n] + add[(size_t)r * D + n] * scale;
    y[i] = v;
    s1 += v;
    s2 += v * v;
  }
#pragma unroll
  for (int o = 16; o > 0; o >>= 1) {
    s1 += __shfl_xor(s1, o, 32);
    s2 += __shfl_xor(s2, o, 32);
  }
  if (lane == 0) { rbuf[w] = s1; rbuf[8 + w] = s2; }
  __syncthreads();
  if (tid == 0) {
    float t1 = 0.0f, t2 = 0.0f;
    for (int i = 0; i < 8; ++i) { t1 += rbuf[i]; t2 += rbuf[8 + i]; }
    float mu = t1 / (float)D;
    float var = t2 / (float)D - mu * mu;
    s_mu = mu;
    s_rstd = rsqrtf(var + LN_EPS);
  }
  __syncthreads();
  float mu = s_mu, rstd = s_rstd;
#pragma unroll
  for (int i = 0; i < 4; ++i) {
    int n = tid + i * 256;
    float v = (y[i] - mu) * rstd * gamma[n] + beta[n];
    if (out_f32) out_f32[(size_t)r * D + n] = v;
    if (out_f16) out_f16[(size_t)r * D + n] = (_Float16)v;
  }
}

// ================= host side =================
extern "C" void kernel_launch(void* const* d_in, const int* in_sizes, int n_in,
                              void* d_out, int out_size, void* d_ws, size_t ws_size,
                              hipStream_t stream) {
  const float* x    = (const float*)d_in[0];
  const float* Wqkv = (const float*)d_in[1];
  const float* bqkv = (const float*)d_in[2];
  const float* Wg   = (const float*)d_in[3];
  const float* bg   = (const float*)d_in[4];
  const float* Wu   = (const float*)d_in[5];
  const float* bu   = (const float*)d_in[6];
  const float* Wd   = (const float*)d_in[7];
  const float* bd   = (const float*)d_in[8];
  const float* g1   = (const float*)d_in[9];
  const float* b1   = (const float*)d_in[10];
  const float* g2   = (const float*)d_in[11];
  const float* b2   = (const float*)d_in[12];
  float* out = (float*)d_out;

  size_t off = 0;
  auto carve = [&](size_t bytes) -> void* {
    void* p = (char*)d_ws + off;
    off += (bytes + 255) & ~(size_t)255;
    return p;
  };
  _Float16* xh      = (_Float16*)carve((size_t)M_TOT * D * 2);
  _Float16* qp      = (_Float16*)carve((size_t)M_TOT * D * 2);
  _Float16* kpT     = (_Float16*)carve((size_t)M_TOT * D * 2);  // [NB][D][S]
  _Float16* vT      = (_Float16*)carve((size_t)M_TOT * D * 2);  // [NB][D][S]
  _Float16* WqkvT_h = (_Float16*)carve((size_t)N_QKV * D * 2);  // [3D][D]
  _Float16* WgT_h   = (_Float16*)carve((size_t)HP * D * 2);     // [HP][D]
  _Float16* WuT_h   = (_Float16*)carve((size_t)HP * D * 2);
  _Float16* WdT_h   = (_Float16*)carve((size_t)D * HP * 2);     // [D][HP]
  float*    bg_p    = (float*)carve((size_t)HP * 4);
  float*    bu_p    = (float*)carve((size_t)HP * 4);
  _Float16* kvT     = (_Float16*)carve((size_t)NB * D * D * 2); // [NB][D(e)][D(d)]
  float*    ksum    = (float*)carve((size_t)NB * D * 4);
  float*    numb    = (float*)carve((size_t)M_TOT * D * 4);     // reused as ffn
  float*    den     = (float*)carve((size_t)M_TOT * 4);
  float*    x1      = (float*)carve((size_t)M_TOT * D * 4);
  _Float16* x1h     = (_Float16*)carve((size_t)M_TOT * D * 2);
  _Float16* hh      = (_Float16*)carve((size_t)M_TOT * HP * 2);

  const int T = 256;
  // conversions / transposed+padded weights
  k_cvt_f16<<<4096, T, 0, stream>>>(x, xh, M_TOT * D);
  k_transpose_pad<<<2048, T, 0, stream>>>(Wqkv, WqkvT_h, D, N_QKV, N_QKV, D, N_QKV);
  k_transpose_pad<<<2048, T, 0, stream>>>(Wg, WgT_h, D, HP, H, D, H);
  k_transpose_pad<<<2048, T, 0, stream>>>(Wu, WuT_h, D, HP, H, D, H);
  k_transpose_pad<<<2048, T, 0, stream>>>(Wd, WdT_h, HP, D, D, H, D);
  k_pad_bias<<<(HP + T - 1) / T, T, 0, stream>>>(bg, bg_p, H, HP);
  k_pad_bias<<<(HP + T - 1) / T, T, 0, stream>>>(bu, bu_p, H, HP);

  // QKV projection + elu feature map (kp, v stored feature-major)
  k_gemm_qkv<<<dim3(N_QKV / 128, M_TOT / 64), T, 0, stream>>>(xh, WqkvT_h, bqkv,
                                                              qp, kpT, vT);
  // kv context per batch: kv = kp^T @ v (output stored as kvT)
  k_gemm_kv<<<dim3(D / 128, D / 64, NB), T, 0, stream>>>(kpT, vT, kvT);
  // k_sum and den
  k_ksum<<<NB * D / 8, T, 0, stream>>>(kpT, ksum);
  // num = qp @ kv
  k_gemm_num<<<dim3(D / 128, M_TOT / 64), T, 0, stream>>>(qp, kvT, numb);
  k_den<<<M_TOT / 8, T, 0, stream>>>(qp, ksum, den);
  // x1 = LN(x + num/(den+eps))
  k_resid_ln<<<M_TOT, T, 0, stream>>>(x, numb, den, g1, b1, x1, x1h);
  // SwiGLU: h = silu(x1@Wg+bg) * (x1@Wu+bu)
  k_gemm_gateup<<<dim3(HP / 128, M_TOT / 64), T, 0, stream>>>(x1h, WgT_h, WuT_h,
                                                              bg_p, bu_p, hh);
  // ffn = h @ Wd + bd   (reuse num buffer)
  k_gemm_down<<<dim3(D / 128, M_TOT / 64), T, 0, stream>>>(hh, WdT_h, bd, numb);
  // out = LN(x1 + ffn)
  k_resid_ln<<<M_TOT, T, 0, stream>>>(x1, numb, nullptr, g2, b2, out, nullptr);
}
